// PaiAutoencoder_82265803588392
// MI455X (gfx1250) — compile-verified, matrix operand order checked by reading, not run on hardware
//
#include <hip/hip_runtime.h>
#include <hip/hip_bf16.h>

#define BB 32
#define KN 9

typedef __attribute__((ext_vector_type(16))) _Float16 v16h;
typedef __attribute__((ext_vector_type(8)))  float    v8f;

// ---------------------------------------------------------------------------
// One-hot matrix -> column index per row.  Exactly one 1.0 per row, so only a
// single thread writes each cols[row]; race-free and deterministic.
// ---------------------------------------------------------------------------
__global__ void k_onehot_cols(const float* __restrict__ Dm, int* __restrict__ cols,
                              int total, int ncols) {
  int i = blockIdx.x * blockDim.x + threadIdx.x;
  if (i < total && Dm[i] > 0.5f) cols[i / ncols] = i % ncols;
}

// Pool/unpool = row gather: Y[b,p,f] = X[b, cols[p], f]
__global__ void k_pool_gather(const float* __restrict__ X, const int* __restrict__ cols,
                              float* __restrict__ Y, int Nout, int Nin, int F) {
  long long t = (long long)blockIdx.x * blockDim.x + threadIdx.x;
  long long total = (long long)BB * Nout * F;
  if (t >= total) return;
  int f = (int)(t % F);
  int p = (int)((t / F) % Nout);
  int b = (int)(t / ((long long)F * Nout));
  Y[t] = X[((long long)b * Nin + cols[p]) * F + f];
}

__global__ void k_pack_f16(const float* __restrict__ src, _Float16* __restrict__ dst, int n) {
  int i = blockIdx.x * blockDim.x + threadIdx.x;
  if (i < n) dst[i] = (_Float16)src[i];
}

// W [Nf, Kd] f32 row-major -> Wp [Np, Kdp] f16, zero padded
__global__ void k_pack_w(const float* __restrict__ W, _Float16* __restrict__ Wp,
                         int Nf, int Np, int Kd, int Kdp) {
  long long i = (long long)blockIdx.x * blockDim.x + threadIdx.x;
  long long total = (long long)Np * Kdp;
  if (i >= total) return;
  int k = (int)(i % Kdp);
  int r = (int)(i / Kdp);
  Wp[i] = (r < Nf && k < Kd) ? (_Float16)W[(long long)r * Kd + k] : (_Float16)0.f;
}

// ---------------------------------------------------------------------------
// FeaSt conv front half: gather neighbors, 9x9 logits, column softmax over the
// first index (faithful to the reference einsum relabeling), weighted agg.
// Emits the f16 A matrix row (length F*9, padded to Kdp) for the WMMA GEMM.
// One thread per (batch, vertex); 81 accumulators live in registers.
// ---------------------------------------------------------------------------
__global__ void k_feast(const float* __restrict__ x, const int* __restrict__ nbr,
                        const float* __restrict__ wm, const float* __restrict__ bm,
                        _Float16* __restrict__ A, int Nlev, int F, int Kdp) {
  int t = blockIdx.x * blockDim.x + threadIdx.x;
  if (t >= BB * Nlev) return;
  int b = t / Nlev;
  int n = t - b * Nlev;
  const float* xb = x + (long long)b * Nlev * F;
  int idx[KN];
#pragma unroll
  for (int k = 0; k < KN; ++k) idx[k] = nbr[n * KN + k];

  float L[KN * KN];
#pragma unroll
  for (int o = 0; o < KN; ++o) {
    float bo = bm[o];
#pragma unroll
    for (int k = 0; k < KN; ++k) L[o * KN + k] = bo;
  }
  // L[o][k] += (xn[f][k] - xn[f][0]) * wm[o][f]
  for (int f = 0; f < F; ++f) {
    float v[KN];
#pragma unroll
    for (int k = 0; k < KN; ++k) v[k] = xb[(long long)idx[k] * F + f];
    float v0 = v[0];
#pragma unroll
    for (int o = 0; o < KN; ++o) {
      float w = wm[o * F + f];
#pragma unroll
      for (int k = 0; k < KN; ++k) L[o * KN + k] = fmaf(v[k] - v0, w, L[o * KN + k]);
    }
  }
  // softmax over the FIRST index for each fixed second index (reference quirk)
#pragma unroll
  for (int c = 0; c < KN; ++c) {
    float mx = L[c];
#pragma unroll
    for (int a = 1; a < KN; ++a) mx = fmaxf(mx, L[a * KN + c]);
    float s = 0.f;
#pragma unroll
    for (int a = 0; a < KN; ++a) {
      float e = expf(L[a * KN + c] - mx);
      L[a * KN + c] = e;
      s += e;
    }
    float inv = 1.f / s;
#pragma unroll
    for (int a = 0; a < KN; ++a) L[a * KN + c] *= inv;
  }
  // agg[f][o] = sum_k xn[f][k] * P[k][o]; write f16 row, zero-pad tail
  _Float16* Ar = A + (long long)t * Kdp;
  for (int f = 0; f < F; ++f) {
    float v[KN];
#pragma unroll
    for (int k = 0; k < KN; ++k) v[k] = xb[(long long)idx[k] * F + f];
#pragma unroll
    for (int o = 0; o < KN; ++o) {
      float a = 0.f;
#pragma unroll
      for (int k = 0; k < KN; ++k) a = fmaf(v[k], L[k * KN + o], a);
      Ar[f * KN + o] = (_Float16)a;
    }
  }
  for (int c = F * KN; c < Kdp; ++c) Ar[c] = (_Float16)0.f;
}

// ---------------------------------------------------------------------------
// Y[M,Nf] = A[M,Kdp](f16) * Wp[Np,Kdp](f16)^T + bias, fused ELU/identity and
// padding-vertex mask.  One wave32 per 16x16 tile; v_wmma_f32_16x16x32_f16.
// A fragment layout (ISA 7.12.2): lane = M%16, lane-half selects K 8-block,
// paired f16 per VGPR; B symmetric with N in place of M; C/D: VGPR r -> M=r+8*half.
// ---------------------------------------------------------------------------
__global__ void k_gemm_wmma(const _Float16* __restrict__ A, const _Float16* __restrict__ Wp,
                            const float* __restrict__ bias, float* __restrict__ Y,
                            int M, int Nf, int Np, int Kdp, int act, int maskN) {
  long long gtid = (long long)blockIdx.x * blockDim.x + threadIdx.x;
  int wave = (int)(gtid >> 5);
  int lane = threadIdx.x & 31;
  int nTiles = Np >> 4;
  int mTiles = M >> 4;
  if (wave >= mTiles * nTiles) return;           // wave-uniform exit: EXEC stays all-1s
  int tm = wave / nTiles;
  int tn = wave - tm * nTiles;
  int r16 = lane & 15;
  int half = lane >> 4;

  const _Float16* Ar = A + (long long)(tm * 16 + r16) * Kdp;
  const _Float16* Wr = Wp + (long long)(tn * 16 + r16) * Kdp;

  v8f acc = {};
  for (int kd = 0; kd < Kdp; kd += 32) {
    v16h av, bv;
#pragma unroll
    for (int j = 0; j < 8; ++j) {
      int kb = ((j < 4) ? (2 * j) : (16 + 2 * (j - 4))) + half * 8 + kd;
      av[2 * j]     = Ar[kb];
      av[2 * j + 1] = Ar[kb + 1];
      bv[2 * j]     = Wr[kb];
      bv[2 * j + 1] = Wr[kb + 1];
    }
    acc = __builtin_amdgcn_wmma_f32_16x16x32_f16(false, av, false, bv,
                                                 (short)0, acc, false, false);
  }
#pragma unroll
  for (int r = 0; r < 8; ++r) {
    int m = tm * 16 + r + 8 * half;
    int n = tn * 16 + r16;
    if (n < Nf) {
      float val = acc[r] + bias[n];
      if (act) val = (val > 0.f) ? val : (expf(val) - 1.f);      // ELU(alpha=1)
      if (maskN && ((m % maskN) == maskN - 1)) val = 0.f;        // padding vertex
      Y[(long long)m * Nf + n] = val;
    }
  }
}

// ---------------------------------------------------------------------------
extern "C" void kernel_launch(void* const* d_in, const int* in_sizes, int n_in,
                              void* d_out, int out_size, void* d_ws, size_t ws_size,
                              hipStream_t stream) {
  (void)in_sizes; (void)n_in; (void)out_size; (void)ws_size;

  const float* x0      = (const float*)d_in[0];
  const int*   nbr[3]  = {(const int*)d_in[1], (const int*)d_in[2], (const int*)d_in[3]};
  const float* Dm[3]   = {(const float*)d_in[4], (const float*)d_in[6], (const float*)d_in[8]};
  const float* Um[3]   = {(const float*)d_in[5], (const float*)d_in[7], (const float*)d_in[9]};
  const float* e_wm[3] = {(const float*)d_in[10], (const float*)d_in[14], (const float*)d_in[18]};
  const float* e_bm[3] = {(const float*)d_in[11], (const float*)d_in[15], (const float*)d_in[19]};
  const float* e_wc[3] = {(const float*)d_in[12], (const float*)d_in[16], (const float*)d_in[20]};
  const float* e_bc[3] = {(const float*)d_in[13], (const float*)d_in[17], (const float*)d_in[21]};
  const float* fce_w   = (const float*)d_in[22];
  const float* fce_b   = (const float*)d_in[23];
  const float* fcd_w   = (const float*)d_in[24];
  const float* fcd_b   = (const float*)d_in[25];
  const float* dwm[3]  = {(const float*)d_in[26], (const float*)d_in[30], (const float*)d_in[34]};
  const float* dbm[3]  = {(const float*)d_in[27], (const float*)d_in[31], (const float*)d_in[35]};
  const float* dwc[3]  = {(const float*)d_in[28], (const float*)d_in[32], (const float*)d_in[36]};
  const float* dbc[3]  = {(const float*)d_in[29], (const float*)d_in[33], (const float*)d_in[37]};

  const int NS[4]  = {5024, 1257, 315, 80};
  const int ENC[4] = {3, 64, 128, 256};
  const int DEC[4] = {256, 128, 64, 3};

  // ---- workspace carving -------------------------------------------------
  char* ws = (char*)d_ws;
  size_t off = 0;
  auto carve = [&](size_t bytes) -> char* {
    char* p = ws + off;
    off += (bytes + 255) & ~(size_t)255;
    return p;
  };
  int* colsD[3];
  int* colsU[3];
  for (int i = 0; i < 3; ++i) {
    colsD[i] = (int*)carve(sizeof(int) * NS[i + 1]);
    colsU[i] = (int*)carve(sizeof(int) * NS[i]);
  }
  _Float16* arenaA = (_Float16*)carve((size_t)160768 * 576 * 2);  // biggest A (d2)
  _Float16* arenaW = (_Float16*)carve((size_t)20480 * 256 * 2);   // biggest W (fce/fcd)
  float*    arenaY = (float*)carve((size_t)160768 * 64 * 4);      // biggest GEMM out
  float*    arenaX = (float*)carve((size_t)160768 * 64 * 4);      // biggest pooled x
  float*    zbuf   = (float*)carve((size_t)BB * 256 * 4);

  auto cdiv = [](long long a, long long b) { return (int)((a + b - 1) / b); };

  // ---- recover one-hot pooling indices ------------------------------------
  for (int i = 0; i < 3; ++i) {
    int tD = NS[i + 1] * NS[i];
    k_onehot_cols<<<cdiv(tD, 256), 256, 0, stream>>>(Dm[i], colsD[i], tD, NS[i]);
    int tU = NS[i] * NS[i + 1];
    k_onehot_cols<<<cdiv(tU, 256), 256, 0, stream>>>(Um[i], colsU[i], tU, NS[i + 1]);
  }

  // ---- encoder -------------------------------------------------------------
  const float* xcur = x0;
  for (int i = 0; i < 3; ++i) {
    int Nlev = NS[i], F = ENC[i], Fo = ENC[i + 1];
    int Kd = F * KN;
    int Kdp = ((Kd + 31) / 32) * 32;
    int M = BB * Nlev;
    k_feast<<<cdiv(M, 128), 128, 0, stream>>>(xcur, nbr[i], e_wm[i], e_bm[i],
                                              arenaA, Nlev, F, Kdp);
    int Np = ((Fo + 15) / 16) * 16;
    k_pack_w<<<cdiv((long long)Np * Kdp, 256), 256, 0, stream>>>(e_wc[i], arenaW, Fo, Np, Kd, Kdp);
    long long waves = (long long)(M / 16) * (Np / 16);
    k_gemm_wmma<<<cdiv(waves * 32, 128), 128, 0, stream>>>(arenaA, arenaW, e_bc[i], arenaY,
                                                           M, Fo, Np, Kdp, /*elu*/1, Nlev);
    k_pool_gather<<<cdiv((long long)BB * NS[i + 1] * Fo, 256), 256, 0, stream>>>(
        arenaY, colsD[i], arenaX, NS[i + 1], Nlev, Fo);
    xcur = arenaX;
  }

  // ---- FC encode: z = x3.flat @ fce_w^T + fce_b ---------------------------
  int nflat = BB * NS[3] * ENC[3];  // 32*80*256 = 655360
  k_pack_f16<<<cdiv(nflat, 256), 256, 0, stream>>>(arenaX, arenaA, nflat);
  k_pack_w<<<cdiv((long long)256 * 20480, 256), 256, 0, stream>>>(fce_w, arenaW, 256, 256, 20480, 20480);
  {
    long long waves = (BB / 16) * (256 / 16);
    k_gemm_wmma<<<cdiv(waves * 32, 128), 128, 0, stream>>>(arenaA, arenaW, fce_b, zbuf,
                                                           BB, 256, 256, 20480, 0, 0);
  }
  // ---- FC decode: xd = z @ fcd_w^T + fcd_b --------------------------------
  k_pack_f16<<<cdiv(BB * 256, 256), 256, 0, stream>>>(zbuf, arenaA, BB * 256);
  k_pack_w<<<cdiv((long long)20480 * 256, 256), 256, 0, stream>>>(fcd_w, arenaW, 20480, 20480, 256, 256);
  {
    long long waves = (BB / 16) * (20480 / 16);
    k_gemm_wmma<<<cdiv(waves * 32, 128), 128, 0, stream>>>(arenaA, arenaW, fcd_b, arenaY,
                                                           BB, 20480, 20480, 256, 0, 0);
  }

  // ---- decoder -------------------------------------------------------------
  for (int i = 0; i < 3; ++i) {
    int lev = 2 - i;
    int Nout = NS[lev], Nin = NS[lev + 1];
    int F = DEC[i], Fo = DEC[i + 1];
    // unpool: arenaY [B, Nin, F] -> arenaX [B, Nout, F]
    k_pool_gather<<<cdiv((long long)BB * Nout * F, 256), 256, 0, stream>>>(
        arenaY, colsU[lev], arenaX, Nout, Nin, F);
    int Kd = F * KN;
    int Kdp = ((Kd + 31) / 32) * 32;
    int M = BB * Nout;
    k_feast<<<cdiv(M, 128), 128, 0, stream>>>(arenaX, nbr[lev], dwm[i], dbm[i],
                                              arenaA, Nout, F, Kdp);
    int Np = ((Fo + 15) / 16) * 16;  // 128, 64, 16 (Fo=3 padded)
    k_pack_w<<<cdiv((long long)Np * Kdp, 256), 256, 0, stream>>>(dwc[i], arenaW, Fo, Np, Kd, Kdp);
    float* Yout = (i == 2) ? (float*)d_out : arenaY;
    int act = (i == 2) ? 0 : 1;  // last dconv uses identity
    long long waves = (long long)(M / 16) * (Np / 16);
    k_gemm_wmma<<<cdiv(waves * 32, 128), 128, 0, stream>>>(arenaA, arenaW, dbc[i], Yout,
                                                           M, Fo, Np, Kdp, act, Nout);
  }
}
